// Decoder_46703474377364
// MI455X (gfx1250) — matrix operand order repaired and checked by reading, not compiled, not run on hardware
//
#include <hip/hip_runtime.h>
#include <hip/hip_bf16.h>
#include <math.h>

typedef __attribute__((ext_vector_type(16))) _Float16 v16h;
typedef __attribute__((ext_vector_type(8)))  _Float16 v8h;
typedef __attribute__((ext_vector_type(8)))  float    v8f;
typedef int v4i_ __attribute__((vector_size(16)));     // matches builtin's int4 pointee

#define Lq 2
#define Bq 128
#define Hq 1024
#define Sq 512
#define Vq 32000
#define Dq 64
#define WIN 129  // 2*D+1

#define BM 128
#define BN 64
#define BK 32
#define AST 40   // LDS row stride (halfs): 80B, 16B-aligned
#define BST 40

#if __has_builtin(__builtin_amdgcn_global_load_async_to_lds_b128) && \
    __has_builtin(__builtin_amdgcn_s_wait_asynccnt)
#define ASYNC_A 1
#else
#define ASYNC_A 0
#endif

__device__ __forceinline__ float sigmoidf_(float x) { return 1.0f / (1.0f + __expf(-x)); }

// ---------------------------------------------------------------------------
// tile fetch/store helpers
// ---------------------------------------------------------------------------
struct AFetch { uint4 x0, x1; };
struct BFetch { float4 f0, f1; };

__device__ __forceinline__ AFetch fetchA(const _Float16* __restrict__ A, int K, int k0, int tid)
{
    int row = tid >> 1, seg = tid & 1;                 // 32B per thread
    const uint4* src = (const uint4*)(A + (size_t)row * K + k0 + seg * 16);
    AFetch r; r.x0 = src[0]; r.x1 = src[1]; return r;
}
__device__ __forceinline__ void storeA(_Float16* Ab, const AFetch& r, int tid)
{
    int row = tid >> 1, seg = tid & 1;
    uint4* dst = (uint4*)(Ab + row * AST + seg * 16);
    dst[0] = r.x0; dst[1] = r.x1;
}

#if ASYNC_A
__device__ __forceinline__ void loadA_async(const _Float16* __restrict__ A, int K, int k0,
                                            _Float16* Ab, int tid)
{
    int row = tid >> 1, seg = tid & 1;
    const _Float16* g = A + (size_t)row * K + k0 + seg * 16;
    _Float16* l = Ab + row * AST + seg * 16;
    __builtin_amdgcn_global_load_async_to_lds_b128(
        (__attribute__((address_space(1))) v4i_*)g,
        (__attribute__((address_space(3))) v4i_*)l, 0, 0);
    __builtin_amdgcn_global_load_async_to_lds_b128(
        (__attribute__((address_space(1))) v4i_*)(g + 8),
        (__attribute__((address_space(3))) v4i_*)(l + 8), 0, 0);
}
#endif

template<int BTRANS>
__device__ __forceinline__ BFetch fetchB(const float* __restrict__ W1, const float* __restrict__ W2,
                                         int Ksplit, int N, int col0, int k0, int tid)
{
    BFetch r;
    if (BTRANS == 0) {
        int n = tid >> 2, q = tid & 3;
        int kg = k0 + q * 8;
        const float* Wp; int kk;
        if (kg < Ksplit) { Wp = W1; kk = kg; } else { Wp = W2; kk = kg - Ksplit; }
        const float4* src = (const float4*)(Wp + (size_t)(col0 + n) * Ksplit + kk);
        r.f0 = src[0]; r.f1 = src[1];
        __builtin_prefetch((const char*)src + BK * 4, 0, 1);
    } else {
        int kk = tid >> 3, nseg = tid & 7;
        const float4* src = (const float4*)(W1 + (size_t)(k0 + kk) * N + col0 + nseg * 8);
        r.f0 = src[0]; r.f1 = src[1];
    }
    return r;
}

template<int BTRANS>
__device__ __forceinline__ void storeB(_Float16* Bb, const BFetch& r, int tid)
{
    float fv[8] = { r.f0.x, r.f0.y, r.f0.z, r.f0.w, r.f1.x, r.f1.y, r.f1.z, r.f1.w };
    if (BTRANS == 0) {
        int n = tid >> 2, q = tid & 3;
        v8h hv;
        #pragma unroll
        for (int i = 0; i < 8; i++) hv[i] = (_Float16)fv[i];
        *(v8h*)(Bb + n * BST + q * 8) = hv;
    } else {
        int kk = tid >> 3, nseg = tid & 7;
        #pragma unroll
        for (int i = 0; i < 8; i++) Bb[(nseg * 8 + i) * BST + kk] = (_Float16)fv[i];
    }
}

// ---------------------------------------------------------------------------
// Generic WMMA GEMM:  C[m,n] = act( sum_k A[m,k] * Welem(n,k) + bias1[n] + bias2[n] )
//   A: [128, K] f16 row-major
//   BTRANS==0: Welem(n,k) = (k<Ksplit ? W1[n*Ksplit+k] : W2[n*Ksplit+(k-Ksplit)])
//   BTRANS==1: Welem(n,k) = W1[k*N + n]   (A @ W)
// Double-buffered LDS, software-pipelined: one barrier per K-tile.
// A tile goes global->LDS via async DMA (ASYNCcnt) when available.
// Grid: N/64 blocks of 256 threads (8 waves); wave = 16 rows x 64 cols.
// ---------------------------------------------------------------------------
template<int TANH, int BTRANS>
__global__ void __launch_bounds__(256)
gemm_wmma_kernel(const _Float16* __restrict__ A,
                 const float* __restrict__ W1,
                 const float* __restrict__ W2,
                 const float* __restrict__ bias1,
                 const float* __restrict__ bias2,
                 float* __restrict__ C,
                 int K, int Ksplit, int N)
{
    __shared__ _Float16 As[2][BM * AST];
    __shared__ _Float16 Bs[2][BN * BST];

    const int tid  = threadIdx.x;
    const int lane = tid & 31;
    const int wv   = tid >> 5;
    const int col0 = blockIdx.x * BN;
    const int rowBase = wv * 16;

    v8f acc[4] = {};

    // ---- prologue: stage tile 0 into buffer 0 ----
#if ASYNC_A
    loadA_async(A, K, 0, As[0], tid);
#else
    { AFetch a0 = fetchA(A, K, 0, tid); storeA(As[0], a0, tid); }
#endif
    { BFetch b0 = fetchB<BTRANS>(W1, W2, Ksplit, N, col0, 0, tid); storeB<BTRANS>(Bs[0], b0, tid); }
#if ASYNC_A
    __builtin_amdgcn_s_wait_asynccnt(0);
#endif
    __syncthreads();

    int p = 0;
    for (int k0 = 0; k0 < K; k0 += BK) {
        const bool more = (k0 + BK) < K;

        // ---- issue global fetch of next tile while computing current ----
        BFetch bnext;
#if ASYNC_A
        if (more) {
            loadA_async(A, K, k0 + BK, As[p ^ 1], tid);
            bnext = fetchB<BTRANS>(W1, W2, Ksplit, N, col0, k0 + BK, tid);
        }
#else
        AFetch anext;
        if (more) {
            anext = fetchA(A, K, k0 + BK, tid);
            bnext = fetchB<BTRANS>(W1, W2, Ksplit, N, col0, k0 + BK, tid);
        }
#endif

        // ---- compute on buffer p: issue all ds_loads, then 4 back-to-back WMMAs
        const _Float16* Ab = As[p];
        const _Float16* Bb = Bs[p];

        const int mh  = lane & 15;
        const int kba = (lane < 16) ? 0 : 8;     // 16-bit A 16x32 layout
        v8h alo = *(const v8h*)(Ab + (rowBase + mh) * AST + kba);
        v8h ahi = *(const v8h*)(Ab + (rowBase + mh) * AST + kba + 16);
        v16h afrag;
        #pragma unroll
        for (int i = 0; i < 8; i++) { afrag[i] = alo[i]; afrag[i + 8] = ahi[i]; }

        const int kbb = (lane < 16) ? 0 : 8;
        v16h bfrag[4];
        #pragma unroll
        for (int ct = 0; ct < 4; ct++) {
            int n = ct * 16 + (lane & 15);
            v8h blo = *(const v8h*)(Bb + n * BST + kbb);
            v8h bhi = *(const v8h*)(Bb + n * BST + kbb + 16);
            #pragma unroll
            for (int i = 0; i < 8; i++) { bfrag[ct][i] = blo[i]; bfrag[ct][i + 8] = bhi[i]; }
        }
        #pragma unroll
        for (int ct = 0; ct < 4; ct++) {
            acc[ct] = __builtin_amdgcn_wmma_f32_16x16x32_f16(
                false, afrag, false, bfrag[ct], (short)0, acc[ct], false, false);
        }

        // ---- commit next tile into the other buffer ----
        if (more) {
#if !ASYNC_A
            storeA(As[p ^ 1], anext, tid);
#endif
            storeB<BTRANS>(Bs[p ^ 1], bnext, tid);
#if ASYNC_A
            __builtin_amdgcn_s_wait_asynccnt(0);
#endif
            __syncthreads();
            p ^= 1;
        }
    }

    // ---- epilogue: C/D 16x16 f32 layout ----
    const int mrow0 = rowBase + ((lane < 16) ? 0 : 8);
    #pragma unroll
    for (int ct = 0; ct < 4; ct++) {
        int n = col0 + ct * 16 + (lane & 15);
        float bv = 0.0f;
        if (bias1) bv += bias1[n];
        if (bias2) bv += bias2[n];
        #pragma unroll
        for (int r = 0; r < 8; r++) {
            int m = mrow0 + r;
            float v = acc[ct][r] + bv;
            if (TANH) v = tanhf(v);
            C[(size_t)m * N + n] = v;
        }
    }
}

// ---------------------------------------------------------------------------
__global__ void f32_to_f16_kernel(const float* __restrict__ src, _Float16* __restrict__ dst, int n)
{
    int i = blockIdx.x * blockDim.x + threadIdx.x;
    if (i < n) dst[i] = (_Float16)src[i];
}

// p_t = S*sigmoid(tanh_y . v_pt + b_v);  s = clip(rint(p_t), D, S-1-D)
__global__ void __launch_bounds__(256)
pt_kernel(const float* __restrict__ tanh_y, const float* __restrict__ v_pt,
          const float* __restrict__ b_v, float* __restrict__ p_t,
          int* __restrict__ s_i, float* __restrict__ s_out)
{
    __shared__ float red[256];
    int b = blockIdx.x, tid = threadIdx.x;
    float acc = 0.0f;
    for (int k = tid; k < Hq; k += 256) acc += tanh_y[(size_t)b * Hq + k] * v_pt[k];
    red[tid] = acc; __syncthreads();
    for (int s = 128; s > 0; s >>= 1) { if (tid < s) red[tid] += red[tid + s]; __syncthreads(); }
    if (tid == 0) {
        float p = (float)Sq * sigmoidf_(red[0] + b_v[0]);
        int s = (int)rintf(p);
        s = min(max(s, Dq), Sq - 1 - Dq);
        p_t[b] = p;
        s_i[b] = s;
        s_out[b] = (float)s;
    }
}

// scores[b,j] = u[b,:] . h_s[s[b]-D+j, b, :] + b_a
__global__ void __launch_bounds__(128)
scores_kernel(const float* __restrict__ u, const float* __restrict__ h_s,
              const int* __restrict__ s_i, const float* __restrict__ b_a,
              float* __restrict__ scores)
{
    __shared__ float red[128];
    int j = blockIdx.x, b = blockIdx.y, tid = threadIdx.x;
    int row = s_i[b] - Dq + j;
    const float* hl = h_s + ((size_t)row * Bq + b) * Hq;
    const float* ub = u + (size_t)b * Hq;
    float acc = 0.0f;
    for (int k = tid; k < Hq; k += 128) acc += ub[k] * hl[k];
    red[tid] = acc; __syncthreads();
    for (int s = 64; s > 0; s >>= 1) { if (tid < s) red[tid] += red[tid + s]; __syncthreads(); }
    if (tid == 0) scores[b * WIN + j] = red[0] + b_a[0];
}

// a_t = softmax(scores)*gauss; context[b,h] = sum_j a_t[j] * h_loc[b,j,h]
__global__ void __launch_bounds__(256)
attn_ctx_kernel(const float* __restrict__ scores, const float* __restrict__ h_s,
                const int* __restrict__ s_i, const float* __restrict__ p_t,
                float* __restrict__ ctx)
{
    __shared__ float sc[WIN];
    __shared__ float at[WIN];
    __shared__ float red[256];
    int b = blockIdx.x, tid = threadIdx.x;
    int sb = s_i[b];
    float p = p_t[b];
    if (tid < WIN) sc[tid] = scores[b * WIN + tid];
    __syncthreads();
    float v = (tid < WIN) ? sc[tid] : -3.4e38f;
    red[tid] = v; __syncthreads();
    for (int s = 128; s > 0; s >>= 1) { if (tid < s) red[tid] = fmaxf(red[tid], red[tid + s]); __syncthreads(); }
    float mx = red[0]; __syncthreads();
    float e = (tid < WIN) ? __expf(sc[tid] - mx) : 0.0f;
    red[tid] = e; __syncthreads();
    for (int s = 128; s > 0; s >>= 1) { if (tid < s) red[tid] += red[tid + s]; __syncthreads(); }
    float inv = 1.0f / red[0];
    if (tid < WIN) {
        float d = (float)(sb - Dq + tid) - p;
        at[tid] = e * inv * __expf(-(d * d) * (1.0f / 2048.0f));   // 2*(D/2)^2 = 2048
    }
    __syncthreads();
    for (int h = tid; h < Hq; h += 256) {
        float acc = 0.0f;
        for (int j = 0; j < WIN; j++)
            acc += at[j] * h_s[((size_t)(sb - Dq + j) * Bq + b) * Hq + h];
        ctx[(size_t)b * Hq + h] = acc;
    }
}

// Acat[b, 0:H] = emb[ids[b]];  Acat[b, H:2H] = h0[b] + ctx[b]
__global__ void prep_l0_kernel(const int* __restrict__ ids, const float* __restrict__ emb,
                               const float* __restrict__ hl, const float* __restrict__ ctx,
                               _Float16* __restrict__ Acat)
{
    int i = blockIdx.x * blockDim.x + threadIdx.x;   // over B*2H
    int b = i >> 11;
    int k = i & 2047;
    float v;
    if (k < Hq) v = emb[(size_t)ids[b] * Hq + k];
    else        v = hl[(size_t)b * Hq + (k - Hq)] + ctx[(size_t)b * Hq + (k - Hq)];
    Acat[i] = (_Float16)v;
}

// Acat[b, 0:H] = x[b];  Acat[b, H:2H] = hl[b] + ctx[b]
__global__ void prep_l_kernel(const float* __restrict__ x, const float* __restrict__ hl,
                              const float* __restrict__ ctx, _Float16* __restrict__ Acat)
{
    int i = blockIdx.x * blockDim.x + threadIdx.x;
    int b = i >> 11;
    int k = i & 2047;
    float v = (k < Hq) ? x[(size_t)b * Hq + k]
                       : hl[(size_t)b * Hq + (k - Hq)] + ctx[(size_t)b * Hq + (k - Hq)];
    Acat[i] = (_Float16)v;
}

__global__ void lstm_cell_kernel(const float* __restrict__ gates, const float* __restrict__ c_in,
                                 float* __restrict__ h_out, float* __restrict__ c_out)
{
    int i = blockIdx.x * blockDim.x + threadIdx.x;   // over B*H
    int b = i >> 10;
    int hx = i & 1023;
    const float* g = gates + (size_t)b * 4 * Hq;
    float gi = g[hx], gf = g[Hq + hx], gg = g[2 * Hq + hx], go = g[3 * Hq + hx];
    float cn = sigmoidf_(gf) * c_in[i] + sigmoidf_(gi) * tanhf(gg);
    float hn = sigmoidf_(go) * tanhf(cn);
    c_out[i] = cn;
    h_out[i] = hn;
}

__global__ void __launch_bounds__(256)
logsoftmax_kernel(float* __restrict__ logits)
{
    __shared__ float red[256];
    int b = blockIdx.x, tid = threadIdx.x;
    float* row = logits + (size_t)b * Vq;
    float mx = -3.4e38f;
    for (int k = tid; k < Vq; k += 256) mx = fmaxf(mx, row[k]);
    red[tid] = mx; __syncthreads();
    for (int s = 128; s > 0; s >>= 1) { if (tid < s) red[tid] = fmaxf(red[tid], red[tid + s]); __syncthreads(); }
    mx = red[0]; __syncthreads();
    float sum = 0.0f;
    for (int k = tid; k < Vq; k += 256) sum += __expf(row[k] - mx);
    red[tid] = sum; __syncthreads();
    for (int s = 128; s > 0; s >>= 1) { if (tid < s) red[tid] += red[tid + s]; __syncthreads(); }
    float lse = mx + __logf(red[0]);
    for (int k = tid; k < Vq; k += 256) row[k] = row[k] - lse;
}

// ---------------------------------------------------------------------------
extern "C" void kernel_launch(void* const* d_in, const int* in_sizes, int n_in,
                              void* d_out, int out_size, void* d_ws, size_t ws_size,
                              hipStream_t stream)
{
    (void)in_sizes; (void)n_in; (void)out_size; (void)ws_size;

    const int*   ids   = (const int*)  d_in[0];
    const float* h     = (const float*)d_in[1];
    const float* c     = (const float*)d_in[2];
    const float* h_s   = (const float*)d_in[3];
    const float* emb   = (const float*)d_in[4];
    const float* W_pt  = (const float*)d_in[5];
    const float* b_pt  = (const float*)d_in[6];
    const float* v_pt  = (const float*)d_in[7];
    const float* b_v   = (const float*)d_in[8];
    const float* W_a   = (const float*)d_in[9];
    const float* b_a   = (const float*)d_in[10];
    const float* W_ih  = (const float*)d_in[11];
    const float* W_hh  = (const float*)d_in[12];
    const float* b_ih  = (const float*)d_in[13];
    const float* b_hh  = (const float*)d_in[14];
    const float* W_out = (const float*)d_in[15];
    const float* b_out = (const float*)d_in[16];

    float* out    = (float*)d_out;
    float* logits = out;                                  // [B, V]
    float* h_out  = out + (size_t)Bq * Vq;                // [L, B, H]
    float* c_out  = h_out + (size_t)Lq * Bq * Hq;         // [L, B, H]
    float* ctx    = c_out + (size_t)Lq * Bq * Hq;         // [B, H]
    float* s_out  = ctx + (size_t)Bq * Hq;                // [B]

    char* ws = (char*)d_ws;
    _Float16* htop16 = (_Float16*)(ws + 0);               // B*H halfs   (256 KB)
    float*    tanh_y = (float*)   (ws + 262144);          // B*H f32     (512 KB)
    float*    u      = (float*)   (ws + 786432);          // B*H f32     (512 KB)
    float*    p_t    = (float*)   (ws + 1310720);         // B f32
    int*      s_i    = (int*)     (ws + 1311232);         // B i32
    float*    scores = (float*)   (ws + 1311744);         // B*129 f32
    _Float16* Acat   = (_Float16*)(ws + 1377792);         // B*2H halfs  (512 KB)
    float*    gates  = (float*)   (ws + 1902080);         // B*4H f32    (2 MB)
    _Float16* x16    = (_Float16*)(ws + 3999232);         // B*H halfs   (256 KB)

    // 1. h_top -> f16
    f32_to_f16_kernel<<<(Bq * Hq) / 256, 256, 0, stream>>>(
        h + (size_t)(Lq - 1) * Bq * Hq, htop16, Bq * Hq);
    // 2. tanh_y = tanh(h_top @ W_pt^T + b_pt)
    gemm_wmma_kernel<1, 0><<<Hq / BN, 256, 0, stream>>>(
        htop16, W_pt, W_pt, b_pt, nullptr, tanh_y, Hq, Hq, Hq);
    // 3. p_t, s
    pt_kernel<<<Bq, 256, 0, stream>>>(tanh_y, v_pt, b_v, p_t, s_i, s_out);
    // 4. u = h_top @ W_a  (transposed weight access)
    gemm_wmma_kernel<0, 1><<<Hq / BN, 256, 0, stream>>>(
        htop16, W_a, W_a, nullptr, nullptr, u, Hq, Hq, Hq);
    // 5. local bilinear scores
    scores_kernel<<<dim3(WIN, Bq), 128, 0, stream>>>(u, h_s, s_i, b_a, scores);
    // 6. softmax * gaussian, context
    attn_ctx_kernel<<<Bq, 256, 0, stream>>>(scores, h_s, s_i, p_t, ctx);
    // 7. layer 0: A = [emb[ids] | h0 + ctx]
    prep_l0_kernel<<<(Bq * 2 * Hq) / 256, 256, 0, stream>>>(ids, emb, h, ctx, Acat);
    // 8. gates0 = A @ [W_ih0 | W_hh0]^T + b_ih0 + b_hh0
    gemm_wmma_kernel<0, 0><<<(4 * Hq) / BN, 256, 0, stream>>>(
        Acat, W_ih, W_hh, b_ih, b_hh, gates, 2 * Hq, Hq, 4 * Hq);
    // 9. cell 0
    lstm_cell_kernel<<<(Bq * Hq) / 256, 256, 0, stream>>>(gates, c, h_out, c_out);
    // 10. layer 1: A = [h_new0 | h1 + ctx]
    prep_l_kernel<<<(Bq * 2 * Hq) / 256, 256, 0, stream>>>(
        h_out, h + (size_t)Bq * Hq, ctx, Acat);
    // 11. gates1
    gemm_wmma_kernel<0, 0><<<(4 * Hq) / BN, 256, 0, stream>>>(
        Acat, W_ih + (size_t)4 * Hq * Hq, W_hh + (size_t)4 * Hq * Hq,
        b_ih + 4 * Hq, b_hh + 4 * Hq, gates, 2 * Hq, Hq, 4 * Hq);
    // 12. cell 1
    lstm_cell_kernel<<<(Bq * Hq) / 256, 256, 0, stream>>>(
        gates, c + (size_t)Bq * Hq, h_out + (size_t)Bq * Hq, c_out + (size_t)Bq * Hq);
    // 13. x -> f16 for output projection
    f32_to_f16_kernel<<<(Bq * Hq) / 256, 256, 0, stream>>>(
        h_out + (size_t)Bq * Hq, x16, Bq * Hq);
    // 14. logits = x @ W_out^T + b_out  (500 blocks, streams 131 MB of weights)
    gemm_wmma_kernel<0, 0><<<Vq / BN, 256, 0, stream>>>(
        x16, W_out, W_out, b_out, nullptr, logits, Hq, Hq, Vq);
    // 15. log_softmax rows
    logsoftmax_kernel<<<Bq, 256, 0, stream>>>(logits);
}
